// MambaModel_84825604096570
// MI455X (gfx1250) — matrix operand order repaired and checked by reading, not compiled, not run on hardware
//
#include <hip/hip_runtime.h>
#include <hip/hip_bf16.h>

// ---------------------------------------------------------------------------
// Mamba forward for MI455X (gfx1250, wave32, WMMA).
// GEMMs: bf16 WMMA (v_wmma_f32_16x16x32_bf16), fp32 accumulate. Each wave
// computes a 16x64 output strip (4 N-tiles): one A fragment feeds 4 WMMAs,
// with explicit double-buffering of fragments so loads for K-step k+32 are
// in flight while the WMMAs for K-step k issue (partial s_wait_loadcnt).
// Selective scan: register-resident recurrence, 2 channels x 16 states per
// wave, cross-lane state reduction via __shfl_xor.
// ---------------------------------------------------------------------------

typedef __attribute__((ext_vector_type(16))) __bf16    v16bf;
typedef __attribute__((ext_vector_type(8)))  float     v8f;
typedef __attribute__((ext_vector_type(4)))  unsigned  v4u;
typedef __attribute__((ext_vector_type(8)))  unsigned  v8u;

#define BB      2
#define LL      1024
#define MDIM    512
#define DINNER  1024
#define DSTATE  16
#define DTRANK  32
#define MVOCAB  32000
#define BL      (BB * LL)

__device__ __forceinline__ unsigned short bf16_rne(float f) {
  unsigned u = __float_as_uint(f);
  u += 0x7FFFu + ((u >> 16) & 1u);       // round-to-nearest-even
  return (unsigned short)(u >> 16);
}

__device__ __forceinline__ unsigned bf16x2_rne(float lo, float hi) {
  return (unsigned)bf16_rne(lo) | ((unsigned)bf16_rne(hi) << 16);
}

__device__ __forceinline__ float siluf(float v) {
  return v / (1.0f + __expf(-v));
}

// ---------------------------------------------------------------------------
// Embedding gather: h[token, :] = embed[x[token], :]
// ---------------------------------------------------------------------------
__global__ void k_embed(const int* __restrict__ x, const float* __restrict__ embed,
                        float* __restrict__ h) {
  int token = blockIdx.x;
  int id = x[token];
  const float* src = embed + (size_t)id * MDIM;
  float* dst = h + (size_t)token * MDIM;
  dst[threadIdx.x]       = src[threadIdx.x];
  dst[threadIdx.x + 256] = src[threadIdx.x + 256];
}

// ---------------------------------------------------------------------------
// fp32 -> bf16 (RNE), 4 elements/thread: b128 load, b64 store
// ---------------------------------------------------------------------------
__global__ void k_f32_to_bf16(const float* __restrict__ in,
                              unsigned* __restrict__ out, int n) {
  int i = (blockIdx.x * blockDim.x + threadIdx.x) * 4;
  if (i >= n) return;
  float4 v = *(const float4*)(in + i);
  out[(i >> 1) + 0] = bf16x2_rne(v.x, v.y);
  out[(i >> 1) + 1] = bf16x2_rne(v.z, v.w);
}

// ---------------------------------------------------------------------------
// RMSNorm over DIM=512, one wave per row, emit bf16 (A-operand for in_proj)
// ---------------------------------------------------------------------------
__global__ void k_rmsnorm_bf16(const float* __restrict__ h, const float* __restrict__ w,
                               unsigned short* __restrict__ out) {
  int wave = threadIdx.x >> 5, lane = threadIdx.x & 31;
  int row = blockIdx.x * 8 + wave;
  const float* hr = h + (size_t)row * MDIM;
  float v[16];
  float sum = 0.f;
#pragma unroll
  for (int j = 0; j < 16; ++j) { v[j] = hr[lane * 16 + j]; sum += v[j] * v[j]; }
#pragma unroll
  for (int m = 16; m >= 1; m >>= 1) sum += __shfl_xor(sum, m, 32);
  float scale = rsqrtf(sum * (1.0f / MDIM) + 1e-5f);
#pragma unroll
  for (int j = 0; j < 16; ++j) {
    int c = lane * 16 + j;
    out[(size_t)row * MDIM + c] = bf16_rne(v[j] * scale * w[c]);
  }
}

// ---------------------------------------------------------------------------
// LayerNorm over DIM=512, one wave per row, emit bf16 (A-operand for head)
// ---------------------------------------------------------------------------
__global__ void k_layernorm_bf16(const float* __restrict__ h, const float* __restrict__ g,
                                 const float* __restrict__ b,
                                 unsigned short* __restrict__ out) {
  int wave = threadIdx.x >> 5, lane = threadIdx.x & 31;
  int row = blockIdx.x * 8 + wave;
  const float* hr = h + (size_t)row * MDIM;
  float v[16];
  float s = 0.f, s2 = 0.f;
#pragma unroll
  for (int j = 0; j < 16; ++j) { v[j] = hr[lane * 16 + j]; s += v[j]; s2 += v[j] * v[j]; }
#pragma unroll
  for (int m = 16; m >= 1; m >>= 1) { s += __shfl_xor(s, m, 32); s2 += __shfl_xor(s2, m, 32); }
  float mu  = s * (1.0f / MDIM);
  float var = s2 * (1.0f / MDIM) - mu * mu;
  float inv = rsqrtf(var + 1e-5f);
#pragma unroll
  for (int j = 0; j < 16; ++j) {
    int c = lane * 16 + j;
    out[(size_t)row * MDIM + c] = bf16_rne((v[j] - mu) * inv * g[c] + b[c]);
  }
}

// ---------------------------------------------------------------------------
// bf16 WMMA GEMM: C[M,N] = A[M,K] @ W[N,K]^T (+bias[n]) (+residual)
// Wave computes a 16x64 strip: one A fragment -> 4 WMMAs per K-step, with
// software pipelining (next K-step's fragments loaded before current WMMAs).
// 8 waves/block span 512 columns; N must be a multiple of 64 (true here).
// A frag: lane holds row (lane&15), K = (lane>>4)*16 + 0..15 -> 2x b128.
// B frag: lane holds col (lane&15), K = (lane>>4)*16 + 0..15 -> contiguous 32B.
// ---------------------------------------------------------------------------
__global__ void __launch_bounds__(256)
k_gemm_bf16(const unsigned short* __restrict__ A, const unsigned short* __restrict__ W,
            const float* __restrict__ bias, const float* __restrict__ residual,
            float* __restrict__ C, int M, int N, int K, int lda, int ldw, int ldc) {
  int wave = threadIdx.x >> 5, lane = threadIdx.x & 31;
  int nblock = blockIdx.x * 8 + wave;      // 64 output columns per wave
  int mtile  = blockIdx.y;
  if (nblock * 64 >= N) return;

  int halfsel = lane >> 4;                 // 0: lanes 0-15, 1: lanes 16-31
  int arow    = mtile * 16 + (lane & 15);
  int kA      = halfsel * 16;
  int kB      = halfsel * 16;

  const unsigned short* ap = A + (size_t)arow * lda;
  const unsigned short* wp = W + (size_t)(nblock * 64 + (lane & 15)) * ldw;
  size_t wstride = (size_t)16 * ldw;       // 16 rows of W per N-tile

  auto load_frags = [&](int kk, v16bf& av, v16bf* bv) {
    v4u a0 = *(const v4u*)(ap + kk + kA);          // K = kk+kA .. +7
    v4u a1 = *(const v4u*)(ap + kk + kA + 8);      // K = kk+kA+8 .. +15
    av = __builtin_bit_cast(v16bf,
         __builtin_shufflevector(a0, a1, 0, 1, 2, 3, 4, 5, 6, 7));
    bv[0] = *(const v16bf*)(wp + 0 * wstride + kk + kB);
    bv[1] = *(const v16bf*)(wp + 1 * wstride + kk + kB);
    bv[2] = *(const v16bf*)(wp + 2 * wstride + kk + kB);
    bv[3] = *(const v16bf*)(wp + 3 * wstride + kk + kB);
  };

  v8f acc0 = {}, acc1 = {}, acc2 = {}, acc3 = {};
  v16bf av, bv[4];
  load_frags(0, av, bv);

  int k = 0;
#pragma unroll 2
  for (; k + 32 < K; k += 32) {
    v16bf avn, bvn[4];
    load_frags(k + 32, avn, bvn);              // next K-step in flight
    __builtin_prefetch(ap + k + 160, 0, 1);    // global_prefetch_b8 (A stream)
    acc0 = __builtin_amdgcn_wmma_f32_16x16x32_bf16(false, av, false, bv[0], (short)0, acc0, false, false);
    acc1 = __builtin_amdgcn_wmma_f32_16x16x32_bf16(false, av, false, bv[1], (short)0, acc1, false, false);
    acc2 = __builtin_amdgcn_wmma_f32_16x16x32_bf16(false, av, false, bv[2], (short)0, acc2, false, false);
    acc3 = __builtin_amdgcn_wmma_f32_16x16x32_bf16(false, av, false, bv[3], (short)0, acc3, false, false);
    av = avn;
    bv[0] = bvn[0]; bv[1] = bvn[1]; bv[2] = bvn[2]; bv[3] = bvn[3];
  }
  acc0 = __builtin_amdgcn_wmma_f32_16x16x32_bf16(false, av, false, bv[0], (short)0, acc0, false, false);
  acc1 = __builtin_amdgcn_wmma_f32_16x16x32_bf16(false, av, false, bv[1], (short)0, acc1, false, false);
  acc2 = __builtin_amdgcn_wmma_f32_16x16x32_bf16(false, av, false, bv[2], (short)0, acc2, false, false);
  acc3 = __builtin_amdgcn_wmma_f32_16x16x32_bf16(false, av, false, bv[3], (short)0, acc3, false, false);

  int mbase = mtile * 16 + halfsel * 8;
  v8f accs[4] = {acc0, acc1, acc2, acc3};
#pragma unroll
  for (int t = 0; t < 4; ++t) {
    int n = nblock * 64 + t * 16 + (lane & 15);
    float bvl = bias ? bias[n] : 0.f;
#pragma unroll
    for (int j = 0; j < 8; ++j) {
      size_t off = (size_t)(mbase + j) * ldc + n;
      float v = accs[t][j] + bvl;
      if (residual) v += residual[off];
      C[off] = v;
    }
  }
}

// ---------------------------------------------------------------------------
// Depthwise causal conv (D_CONV=4) over xin half of xz, + bias, + SiLU.
// Emits fp32 (for scan) and bf16 (A-operand for x_proj GEMM).
// ---------------------------------------------------------------------------
__global__ void k_conv_silu(const float* __restrict__ xz, const float* __restrict__ cw,
                            const float* __restrict__ cb, float* __restrict__ xc,
                            unsigned short* __restrict__ xcb) {
  int idx = blockIdx.x * blockDim.x + threadIdx.x;
  if (idx >= BL * DINNER) return;
  int c = idx & (DINNER - 1);
  int token = idx >> 10;               // DINNER == 1024
  int l = token & (LL - 1);
  float acc = cb[c];
#pragma unroll
  for (int t = 0; t < 4; ++t) {
    int ls = l - 3 + t;
    if (ls >= 0) acc += cw[c * 4 + t] * xz[(size_t)(token - 3 + t) * (2 * DINNER) + c];
  }
  float s = siluf(acc);
  xc[idx] = s;
  xcb[idx] = bf16_rne(s);
}

// ---------------------------------------------------------------------------
// delta[token,d] = softplus(dbc[token,0:32] . dt_w[d,:] + dt_b[d])
// ---------------------------------------------------------------------------
__global__ void k_delta(const float* __restrict__ dbc, const float* __restrict__ dtw,
                        const float* __restrict__ dtb, float* __restrict__ dlt) {
  int idx = blockIdx.x * blockDim.x + threadIdx.x;
  if (idx >= BL * DINNER) return;
  int d = idx & (DINNER - 1);
  int token = idx >> 10;
  const float* row = dbc + (size_t)token * 64;
  float acc = dtb[d];
#pragma unroll
  for (int r = 0; r < DTRANK; ++r) acc += row[r] * dtw[d * DTRANK + r];
  dlt[idx] = (acc > 20.f) ? acc : log1pf(expf(acc));
}

// ---------------------------------------------------------------------------
// Selective scan. Wave = 2 channels x 16 states. Recurrent h in a register.
// Per step: h = exp(delta*A)*h + delta*x*B[s]; y += h*C[s]  (shfl reduce s),
// fused gate silu(z) and D*x; emits bf16 (A-operand for out_proj GEMM).
// ---------------------------------------------------------------------------
__global__ void k_scan(const float* __restrict__ dlt, const float* __restrict__ xc,
                       const float* __restrict__ dbc, const float* __restrict__ A_log,
                       const float* __restrict__ Dp, const float* __restrict__ xz,
                       unsigned short* __restrict__ ybf) {
  int wave = threadIdx.x >> 5, lane = threadIdx.x & 31;
  int b = blockIdx.x / (DINNER / 16);
  int dbase = (blockIdx.x % (DINNER / 16)) * 16;
  int d = dbase + wave * 2 + (lane >> 4);
  int s = lane & 15;

  float Av = -__expf(A_log[(size_t)d * DSTATE + s]);
  float Dv = Dp[d];
  float h = 0.f;

  for (int l = 0; l < LL; ++l) {
    int token = b * LL + l;
    float dl = dlt[(size_t)token * DINNER + d];
    float xv = xc[(size_t)token * DINNER + d];
    float Bm = dbc[(size_t)token * 64 + 32 + s];
    float Cm = dbc[(size_t)token * 64 + 48 + s];
    float dA = __expf(dl * Av);
    h = dA * h + (dl * xv) * Bm;
    float val = h * Cm;
    val += __shfl_xor(val, 1, 32);
    val += __shfl_xor(val, 2, 32);
    val += __shfl_xor(val, 4, 32);
    val += __shfl_xor(val, 8, 32);
    if (s == 0) {
      float y = val + Dv * xv;
      float z = xz[(size_t)token * (2 * DINNER) + DINNER + d];
      ybf[(size_t)token * DINNER + d] = bf16_rne(y * siluf(z));
    }
  }
}

// ---------------------------------------------------------------------------
// Host orchestration (graph-capture safe: only launches on `stream`).
// ---------------------------------------------------------------------------
extern "C" void kernel_launch(void* const* d_in, const int* in_sizes, int n_in,
                              void* d_out, int out_size, void* d_ws, size_t ws_size,
                              hipStream_t stream) {
  (void)in_sizes; (void)n_in; (void)out_size; (void)ws_size;
  const int*   x       = (const int*)  d_in[0];
  const float* embed   = (const float*)d_in[1];
  const float* rms_w   = (const float*)d_in[2];
  const float* in_w    = (const float*)d_in[3];
  const float* conv_w  = (const float*)d_in[4];
  const float* conv_b  = (const float*)d_in[5];
  const float* xproj_w = (const float*)d_in[6];
  const float* dt_w    = (const float*)d_in[7];
  const float* dt_b    = (const float*)d_in[8];
  const float* A_log   = (const float*)d_in[9];
  const float* Dp      = (const float*)d_in[10];
  const float* out_w   = (const float*)d_in[11];
  const float* ln_g    = (const float*)d_in[12];
  const float* ln_b    = (const float*)d_in[13];
  const float* head_w  = (const float*)d_in[14];
  const float* head_b  = (const float*)d_in[15];
  float* out = (float*)d_out;

  size_t off = 0;
  auto alloc = [&](size_t bytes) -> void* {
    void* p = (void*)((char*)d_ws + off);
    off += (bytes + 255) & ~(size_t)255;
    return p;
  };
  float* h   = (float*)alloc(sizeof(float) * BL * MDIM);
  float* xz  = (float*)alloc(sizeof(float) * BL * 2 * DINNER);
  float* xc  = (float*)alloc(sizeof(float) * BL * DINNER);
  float* dbc = (float*)alloc(sizeof(float) * BL * 64);
  float* dlt = (float*)alloc(sizeof(float) * BL * DINNER);
  unsigned short* r_bf      = (unsigned short*)alloc(2 * (size_t)BL * MDIM);
  unsigned short* xc_bf     = (unsigned short*)alloc(2 * (size_t)BL * DINNER);
  unsigned short* y_bf      = (unsigned short*)alloc(2 * (size_t)BL * DINNER);
  unsigned short* hn_bf     = (unsigned short*)alloc(2 * (size_t)BL * MDIM);
  unsigned short* w_in_bf   = (unsigned short*)alloc(2 * (size_t)2 * DINNER * MDIM);
  unsigned short* w_xp_bf   = (unsigned short*)alloc(2 * (size_t)64 * DINNER);
  unsigned short* w_out_bf  = (unsigned short*)alloc(2 * (size_t)MDIM * DINNER);
  unsigned short* w_head_bf = (unsigned short*)alloc(2 * (size_t)MVOCAB * MDIM);

  k_embed<<<BL, 256, 0, stream>>>(x, embed, h);
  { int n = MVOCAB * MDIM;
    k_f32_to_bf16<<<(n / 4 + 255) / 256, 256, 0, stream>>>(head_w, (unsigned*)w_head_bf, n); }

  for (int l = 0; l < 2; ++l) {
    const float* rw = rms_w   + (size_t)l * MDIM;
    const float* iw = in_w    + (size_t)l * 2 * DINNER * MDIM;
    const float* cw = conv_w  + (size_t)l * DINNER * 4;
    const float* cb = conv_b  + (size_t)l * DINNER;
    const float* xw = xproj_w + (size_t)l * 64 * DINNER;
    const float* dw = dt_w    + (size_t)l * DINNER * DTRANK;
    const float* db = dt_b    + (size_t)l * DINNER;
    const float* Al = A_log   + (size_t)l * DINNER * DSTATE;
    const float* Dv = Dp      + (size_t)l * DINNER;
    const float* ow = out_w   + (size_t)l * MDIM * DINNER;

    k_rmsnorm_bf16<<<BL / 8, 256, 0, stream>>>(h, rw, r_bf);

    { int n = 2 * DINNER * MDIM;
      k_f32_to_bf16<<<(n / 4 + 255) / 256, 256, 0, stream>>>(iw, (unsigned*)w_in_bf, n); }
    { dim3 g((2 * DINNER + 511) / 512, BL / 16);
      k_gemm_bf16<<<g, 256, 0, stream>>>(r_bf, w_in_bf, nullptr, nullptr, xz,
                                         BL, 2 * DINNER, MDIM, MDIM, MDIM, 2 * DINNER); }

    { int n = BL * DINNER;
      k_conv_silu<<<(n + 255) / 256, 256, 0, stream>>>(xz, cw, cb, xc, xc_bf); }

    { int n = 64 * DINNER;
      k_f32_to_bf16<<<(n / 4 + 255) / 256, 256, 0, stream>>>(xw, (unsigned*)w_xp_bf, n); }
    { dim3 g(1, BL / 16);   // N=64 -> wave 0 of each block active
      k_gemm_bf16<<<g, 256, 0, stream>>>(xc_bf, w_xp_bf, nullptr, nullptr, dbc,
                                         BL, 64, DINNER, DINNER, DINNER, 64); }

    { int n = BL * DINNER;
      k_delta<<<(n + 255) / 256, 256, 0, stream>>>(dbc, dw, db, dlt); }

    k_scan<<<BB * (DINNER / 16), 256, 0, stream>>>(dlt, xc, dbc, Al, Dv, xz, y_bf);

    { int n = MDIM * DINNER;
      k_f32_to_bf16<<<(n / 4 + 255) / 256, 256, 0, stream>>>(ow, (unsigned*)w_out_bf, n); }
    { dim3 g((MDIM + 511) / 512, BL / 16);
      k_gemm_bf16<<<g, 256, 0, stream>>>(y_bf, w_out_bf, nullptr, h, h,
                                         BL, MDIM, DINNER, DINNER, DINNER, MDIM); }
  }

  k_layernorm_bf16<<<BL / 8, 256, 0, stream>>>(h, ln_g, ln_b, hn_bf);
  { dim3 g((MVOCAB + 511) / 512, BL / 16);
    k_gemm_bf16<<<g, 256, 0, stream>>>(hn_bf, w_head_bf, head_b, nullptr, out,
                                       BL, MVOCAB, MDIM, MDIM, MDIM, MVOCAB); }
}
